// ScaledDotProductAttentionBlock_13520557047958
// MI455X (gfx1250) — compile-verified, matrix operand order checked by reading, not run on hardware
//
#include <hip/hip_runtime.h>
#include <hip/hip_fp16.h>

typedef __attribute__((ext_vector_type(16))) _Float16 v16h;
typedef __attribute__((ext_vector_type(8)))  _Float16 v8h;
typedef __attribute__((ext_vector_type(2)))  _Float16 v2h;
typedef __attribute__((ext_vector_type(8)))  float    v8f;
typedef __attribute__((ext_vector_type(4)))  float    f4;

#define B_      2
#define HEADS_  16
#define SEQ_    2048
#define DMODEL_ 1024
#define DH_     64
#define KB_     32     // keys per mainloop iteration
#define QB_     16     // q rows per tile
#define QT_     2      // q tiles per wave (32 q rows / wave)
#define WAVES_  4      // waves per workgroup (share one (b,h))
#define NKB_    (SEQ_ / KB_)

// scale = (1/sqrt(64)) * log2(e): scores come out of WMMA already in log2 domain
#define QSCALE_ 0.18033688011112042f

#if __has_builtin(__builtin_amdgcn_exp2f)
#define EXP2F(x) __builtin_amdgcn_exp2f(x)
#else
#define EXP2F(x) exp2f(x)
#endif

// Load 16 f16 elements in WMMA A/B fragment order from a row-major f16 row:
// elements 0..7 = base[ko..ko+7], elements 8..15 = base[ko+16..ko+23]
static __device__ __forceinline__ v16h ld_frag(const _Float16* base, int ko) {
    v8h lo = *(const v8h*)(base + ko);
    v8h hi = *(const v8h*)(base + ko + 16);
    v16h r;
#pragma unroll
    for (int j = 0; j < 8; ++j) { r[j] = lo[j]; r[8 + j] = hi[j]; }
    return r;
}

// raw v_max_num_f32 (no sNaN-canonicalize ops the compiler adds around fmaxf)
static __device__ __forceinline__ float vmax_raw(float a, float b) {
    float d;
    asm("v_max_num_f32 %0, %1, %2" : "=v"(d) : "v"(a), "v"(b));
    return d;
}

// ---- DPP16 16-lane butterfly reductions (lanes 0-15 and 16-31 reduce independently) ----
template <int CTRL>
static __device__ __forceinline__ float dpp_mov(float x) {
    return __int_as_float(
        __builtin_amdgcn_mov_dpp(__float_as_int(x), CTRL, 0xF, 0xF, true));
}
static __device__ __forceinline__ float redmax16(float t) {
    t = vmax_raw(t, dpp_mov<0xB1>(t));    // quad_perm(1,0,3,2)
    t = vmax_raw(t, dpp_mov<0x4E>(t));    // quad_perm(2,3,0,1)
    t = vmax_raw(t, dpp_mov<0x141>(t));   // row_half_mirror
    t = vmax_raw(t, dpp_mov<0x140>(t));   // row_mirror
    return t;
}
static __device__ __forceinline__ float redsum16(float t) {
    t += dpp_mov<0xB1>(t);
    t += dpp_mov<0x4E>(t);
    t += dpp_mov<0x141>(t);
    t += dpp_mov<0x140>(t);
    return t;
}

// ---- staging registers: K rows are (1 row x 16 cols), V is (2 rows x 8 cols) ----
struct Stage {
    f4 k[4];           // K: row srow, cols scs..scs+15
    f4 va[2], vb[2];   // V: rows vkp, vkp+1, cols vcs..vcs+7
};

static __device__ __forceinline__ void stage_load(const float* kp, const float* vp,
                                                  Stage& st) {
#pragma unroll
    for (int j = 0; j < 4; ++j) st.k[j] = *(const f4*)(kp + 4 * j);
    st.va[0] = *(const f4*)(vp);
    st.va[1] = *(const f4*)(vp + 4);
    st.vb[0] = *(const f4*)(vp + DMODEL_);
    st.vb[1] = *(const f4*)(vp + DMODEL_ + 4);
}

static __device__ __forceinline__ void stage_commit(_Float16* Kbuf, _Float16* Vbuf,
                                                    int srow, int scs, int vkp, int vcs,
                                                    const Stage& st) {
    v8h ka, kb2;
#pragma unroll
    for (int j = 0; j < 4; ++j) {
        ka[j]      = (_Float16)st.k[0][j];  ka[4 + j]  = (_Float16)st.k[1][j];
        kb2[j]     = (_Float16)st.k[2][j];  kb2[4 + j] = (_Float16)st.k[3][j];
    }
    *(v8h*)&Kbuf[srow * DH_ + scs]     = ka;
    *(v8h*)&Kbuf[srow * DH_ + scs + 8] = kb2;
    // V transposed: pack (key vkp, key vkp+1) pairs -> one b32 store per d-column
#pragma unroll
    for (int hh = 0; hh < 2; ++hh)
#pragma unroll
        for (int c = 0; c < 4; ++c) {
            v2h p;
            p[0] = (_Float16)st.va[hh][c];
            p[1] = (_Float16)st.vb[hh][c];
            *(v2h*)&Vbuf[(vcs + 4 * hh + c) * KB_ + vkp] = p;
        }
}

__global__ __launch_bounds__(WAVES_ * 32, 2)
void fa_fwd_f16wmma(const float* __restrict__ Q, const float* __restrict__ K,
                    const float* __restrict__ V, float* __restrict__ O)
{
    __shared__ __align__(16) _Float16 Ksh [2][KB_ * DH_];             // 32x64 (key,d), dbl-buffered
    __shared__ __align__(16) _Float16 VTsh[2][DH_ * KB_];             // 64x32 (d,key), dbl-buffered
    __shared__ __align__(16) _Float16 Psh[WAVES_][QT_][QB_ * KB_];    // per-wave/tile 16x32 (q,key)

    const int tid  = threadIdx.x;
    const int wave = tid >> 5;
    const int lane = tid & 31;
    const int ln   = lane & 15;
    const int ko   = (lane < 16) ? 0 : 8;

    const int bh     = blockIdx.x >> 4;     // 0..31 : (b,h)
    const int qchunk = blockIdx.x & 15;     // 0..15
    const int b      = bh >> 4;
    const int h      = bh & (HEADS_ - 1);
    const int q0     = (qchunk * WAVES_ + wave) * (QB_ * QT_);

    const size_t headoff = (size_t)b * SEQ_ * DMODEL_ + (size_t)h * DH_;
    const float* Qh = Q + headoff;
    const float* Kh = K + headoff;
    const float* Vh = V + headoff;
    float*       Oh = O + headoff;

    // staging coordinates
    const int srow = tid >> 2;           // K: 0..31 key row
    const int scs  = (tid & 3) * 16;     // K: col segment
    const int vkp  = (tid & 15) * 2;     // V: key row pair base
    const int vcs  = (tid >> 4) * 8;     // V: col segment (8 cols)
    const float* Kst = Kh + (size_t)srow * DMODEL_ + scs;
    const float* Vst = Vh + (size_t)vkp * DMODEL_ + vcs;

    // ---- Q tiles (2 x 16 x 64) -> A fragments, scale*log2(e) folded in ----
    v16h aq[QT_][2];
#pragma unroll
    for (int qt = 0; qt < QT_; ++qt) {
        const float* qrow = Qh + (size_t)(q0 + qt * QB_ + ln) * DMODEL_;
#pragma unroll
        for (int kc = 0; kc < 2; ++kc) {
            const float* p = qrow + kc * 32 + ko;
            f4 x0 = *(const f4*)(p);
            f4 x1 = *(const f4*)(p + 4);
            f4 x2 = *(const f4*)(p + 16);
            f4 x3 = *(const f4*)(p + 20);
            v16h a;
#pragma unroll
            for (int j = 0; j < 4; ++j) {
                a[j]      = (_Float16)(x0[j] * QSCALE_);
                a[4 + j]  = (_Float16)(x1[j] * QSCALE_);
                a[8 + j]  = (_Float16)(x2[j] * QSCALE_);
                a[12 + j] = (_Float16)(x3[j] * QSCALE_);
            }
            aq[qt][kc] = a;
        }
    }

    // ---- accumulators / online-softmax state (log2 domain) ----
    v8f o[QT_][4];
#pragma unroll
    for (int qt = 0; qt < QT_; ++qt)
#pragma unroll
        for (int nt = 0; nt < 4; ++nt)
#pragma unroll
            for (int r = 0; r < 8; ++r) o[qt][nt][r] = 0.0f;

    float mr[QT_][8], lr[QT_][8];
#pragma unroll
    for (int qt = 0; qt < QT_; ++qt)
#pragma unroll
        for (int r = 0; r < 8; ++r) { mr[qt][r] = -1.0e30f; lr[qt][r] = 0.0f; }

    // ---- prologue: stage chunk 0 into buffer 0 ----
    {
        Stage st;
        stage_load(Kst, Vst, st);
        stage_commit(Ksh[0], VTsh[0], srow, scs, vkp, vcs, st);
    }
    __syncthreads();

    for (int kb = 0; kb < NKB_; ++kb) {
        const int cur   = kb & 1;
        const bool more = (kb + 1) < NKB_;

        // ---- issue next chunk's global loads early (latency hidden under WMMAs) ----
        Stage st;
        if (more) {
            const float* kn = Kst + (size_t)(kb + 1) * KB_ * DMODEL_;
            const float* vn = Vst + (size_t)(kb + 1) * KB_ * DMODEL_;
            stage_load(kn, vn, st);
            if (kb + 2 < NKB_) {
                __builtin_prefetch(kn + (size_t)KB_ * DMODEL_, 0, 1);
                __builtin_prefetch(vn + (size_t)KB_ * DMODEL_, 0, 1);
            }
        }

        // ---- S = Q'Kt (log2-scaled) : 2 q-tiles x 2 key-tiles, K-dim 64 = 2 chunks ----
        const _Float16* Kc  = Ksh[cur];
        const _Float16* k0p = &Kc[ln * DH_];
        const _Float16* k1p = &Kc[(16 + ln) * DH_];
        v16h b00 = ld_frag(k0p, ko);
        v16h b01 = ld_frag(k0p + 32, ko);
        v16h b10 = ld_frag(k1p, ko);
        v16h b11 = ld_frag(k1p + 32, ko);

        v8f sc[QT_][2];
#pragma unroll
        for (int qt = 0; qt < QT_; ++qt)
#pragma unroll
            for (int nt = 0; nt < 2; ++nt)
#pragma unroll
                for (int r = 0; r < 8; ++r) sc[qt][nt][r] = 0.0f;

        // two independent WMMA chains (one per q-tile), shared B fragments
#pragma unroll
        for (int qt = 0; qt < QT_; ++qt) {
            sc[qt][0] = __builtin_amdgcn_wmma_f32_16x16x32_f16(false, aq[qt][0], false, b00, (short)0, sc[qt][0], false, false);
            sc[qt][1] = __builtin_amdgcn_wmma_f32_16x16x32_f16(false, aq[qt][0], false, b10, (short)0, sc[qt][1], false, false);
            sc[qt][0] = __builtin_amdgcn_wmma_f32_16x16x32_f16(false, aq[qt][1], false, b01, (short)0, sc[qt][0], false, false);
            sc[qt][1] = __builtin_amdgcn_wmma_f32_16x16x32_f16(false, aq[qt][1], false, b11, (short)0, sc[qt][1], false, false);
        }

        // ---- online softmax in log2 domain (DPP 16-lane reductions, raw maxes) ----
#pragma unroll
        for (int qt = 0; qt < QT_; ++qt) {
            float corr[8];
#pragma unroll
            for (int r = 0; r < 8; ++r) {
                float v0 = sc[qt][0][r];
                float v1 = sc[qt][1][r];
                float t  = redmax16(vmax_raw(v0, v1));
                float mn = vmax_raw(mr[qt][r], t);
                float c  = EXP2F(mr[qt][r] - mn);
                mr[qt][r] = mn;
                float p0 = EXP2F(v0 - mn);
                float p1 = EXP2F(v1 - mn);
                sc[qt][0][r] = p0; sc[qt][1][r] = p1;
                float rs = redsum16(p0 + p1);
                lr[qt][r] = lr[qt][r] * c + rs;
                corr[r] = c;
            }
#pragma unroll
            for (int nt = 0; nt < 4; ++nt)
#pragma unroll
                for (int r = 0; r < 8; ++r) o[qt][nt][r] *= corr[r];
        }

        // ---- P: C-layout -> LDS -> A-layout (same wave, DS in-order) ----
#pragma unroll
        for (int qt = 0; qt < QT_; ++qt) {
            _Float16* pp = &Psh[wave][qt][0];
            const int rbase = (lane < 16) ? 0 : 8;
#pragma unroll
            for (int r = 0; r < 8; ++r) {
                pp[(rbase + r) * KB_ + ln]      = (_Float16)sc[qt][0][r];
                pp[(rbase + r) * KB_ + 16 + ln] = (_Float16)sc[qt][1][r];
            }
        }
        v16h ap0 = ld_frag(&Psh[wave][0][ln * KB_], ko);
        v16h ap1 = ld_frag(&Psh[wave][1][ln * KB_], ko);

        // ---- O += P * V : 2 q-tiles x 4 dh tiles, K-dim = 32 keys, shared B ----
        const _Float16* Vc = VTsh[cur];
        v16h bv0 = ld_frag(&Vc[(0 * 16 + ln) * KB_], ko);
        v16h bv1 = ld_frag(&Vc[(1 * 16 + ln) * KB_], ko);
        v16h bv2 = ld_frag(&Vc[(2 * 16 + ln) * KB_], ko);
        v16h bv3 = ld_frag(&Vc[(3 * 16 + ln) * KB_], ko);
        o[0][0] = __builtin_amdgcn_wmma_f32_16x16x32_f16(false, ap0, false, bv0, (short)0, o[0][0], false, false);
        o[1][0] = __builtin_amdgcn_wmma_f32_16x16x32_f16(false, ap1, false, bv0, (short)0, o[1][0], false, false);
        o[0][1] = __builtin_amdgcn_wmma_f32_16x16x32_f16(false, ap0, false, bv1, (short)0, o[0][1], false, false);
        o[1][1] = __builtin_amdgcn_wmma_f32_16x16x32_f16(false, ap1, false, bv1, (short)0, o[1][1], false, false);
        o[0][2] = __builtin_amdgcn_wmma_f32_16x16x32_f16(false, ap0, false, bv2, (short)0, o[0][2], false, false);
        o[1][2] = __builtin_amdgcn_wmma_f32_16x16x32_f16(false, ap1, false, bv2, (short)0, o[1][2], false, false);
        o[0][3] = __builtin_amdgcn_wmma_f32_16x16x32_f16(false, ap0, false, bv3, (short)0, o[0][3], false, false);
        o[1][3] = __builtin_amdgcn_wmma_f32_16x16x32_f16(false, ap1, false, bv3, (short)0, o[1][3], false, false);

        // ---- commit next chunk to the other buffer, then one barrier ----
        if (more)
            stage_commit(Ksh[cur ^ 1], VTsh[cur ^ 1], srow, scs, vkp, vcs, st);
        __syncthreads();
    }

    // ---- epilogue: normalize and store (C-layout scatter) ----
#pragma unroll
    for (int qt = 0; qt < QT_; ++qt)
#pragma unroll
        for (int r = 0; r < 8; ++r) {
            const int row = qt * QB_ + r + ((lane < 16) ? 0 : 8);
            const float inv = 1.0f / lr[qt][r];
            float* orow = Oh + (size_t)(q0 + row) * DMODEL_;
#pragma unroll
            for (int nt = 0; nt < 4; ++nt)
                orow[nt * 16 + ln] = o[qt][nt][r] * inv;
        }
}

extern "C" void kernel_launch(void* const* d_in, const int* in_sizes, int n_in,
                              void* d_out, int out_size, void* d_ws, size_t ws_size,
                              hipStream_t stream) {
    (void)in_sizes; (void)n_in; (void)d_ws; (void)ws_size; (void)out_size;
    const float* q = (const float*)d_in[0];
    const float* k = (const float*)d_in[1];
    const float* v = (const float*)d_in[2];
    float* out = (float*)d_out;
    // grid: 32 (b,h) pairs * 16 q-chunks; block: 4 waves of 32 (each wave: 32 q rows)
    dim3 grid(B_ * HEADS_ * (SEQ_ / (WAVES_ * QB_ * QT_)));
    dim3 block(WAVES_ * 32);
    fa_fwd_f16wmma<<<grid, block, 0, stream>>>(q, k, v, out);
}
